// Attention_11690900980356
// MI455X (gfx1250) — compile-verified, matrix operand order checked by reading, not run on hardware
//
#include <hip/hip_runtime.h>

// ---------------------------------------------------------------------------
// Types for CDNA5 WMMA
// ---------------------------------------------------------------------------
typedef __attribute__((ext_vector_type(16))) __bf16         v16bf;
typedef __attribute__((ext_vector_type(8)))  float          v8f;
typedef __attribute__((ext_vector_type(8)))  unsigned short ush8;
typedef __attribute__((ext_vector_type(4)))  unsigned short ush4;
typedef int v4i __attribute__((vector_size(16)));   // matches builtin pointee

__device__ __forceinline__ unsigned short f32_to_bf16(float f) {
    unsigned int u = __builtin_bit_cast(unsigned int, f);
    u += 0x7fffu + ((u >> 16) & 1u);      // round-to-nearest-even
    return (unsigned short)(u >> 16);
}

// ---------------------------------------------------------------------------
// CDNA5 async global->LDS copy (ASYNCcnt path), guarded so the file still
// compiles on toolchains that lack the builtin (falls back to reg staging).
//   void __builtin_amdgcn_global_load_async_to_lds_b128(
//        global v4i* src, local v4i* dst, imm int offset, imm int cpol)
// ---------------------------------------------------------------------------
#if __has_builtin(__builtin_amdgcn_global_load_async_to_lds_b128)
#define HAVE_ASYNC_LDS 1
__device__ __forceinline__ void async_ld16(const unsigned short* g, unsigned short* l) {
    __builtin_amdgcn_global_load_async_to_lds_b128(
        (__attribute__((address_space(1))) v4i*)(unsigned long long)g,
        (__attribute__((address_space(3))) v4i*)(unsigned int)(unsigned long long)l,
        0, 0);
}
#if __has_builtin(__builtin_amdgcn_s_wait_asynccnt)
#define WAIT_ASYNC(n) __builtin_amdgcn_s_wait_asynccnt(n)
#else
#define WAIT_ASYNC(n) asm volatile("s_wait_asynccnt %0" ::"i"(n) : "memory")
#endif
#else
#define HAVE_ASYNC_LDS 0
#endif

// ---------------------------------------------------------------------------
// f32 -> bf16 bulk convert (4 elems / thread, float4 loads)
// ---------------------------------------------------------------------------
__global__ __launch_bounds__(256)
void cvt_f32_bf16(const float* __restrict__ src, unsigned short* __restrict__ dst,
                  long long n) {
    long long i = ((long long)blockIdx.x * 256 + threadIdx.x) * 4;
    if (i + 3 < n) {
        float4 v = *(const float4*)(src + i);
        dst[i + 0] = f32_to_bf16(v.x);
        dst[i + 1] = f32_to_bf16(v.y);
        dst[i + 2] = f32_to_bf16(v.z);
        dst[i + 3] = f32_to_bf16(v.w);
    }
}

// ---------------------------------------------------------------------------
// Tiled bf16 transpose through LDS: out[c][r] = in[r][c], per batch (grid.z).
// ---------------------------------------------------------------------------
__global__ __launch_bounds__(256)
void transpose_bf16(const unsigned short* __restrict__ in,
                    unsigned short* __restrict__ out, int rows, int cols) {
    __shared__ __align__(8) unsigned short t[32][33];
    const long long base = (long long)blockIdx.z * rows * cols;
    const int r0 = blockIdx.y * 32, c0 = blockIdx.x * 32;
    {
        const int r  = threadIdx.x >> 3;
        const int c4 = (threadIdx.x & 7) * 4;
        ush4 v = *(const ush4*)&in[base + (long long)(r0 + r) * cols + c0 + c4];
#pragma unroll
        for (int j = 0; j < 4; ++j) t[r][c4 + j] = v[j];
    }
    __syncthreads();
    {
        const int c  = threadIdx.x >> 3;
        const int r4 = (threadIdx.x & 7) * 4;
        ush4 w;
#pragma unroll
        for (int j = 0; j < 4; ++j) w[j] = t[r4 + j][c];
        *(ush4*)&out[base + (long long)(c0 + c) * rows + r0 + r4] = w;
    }
}

// ---------------------------------------------------------------------------
// Tiled bf16 WMMA GEMM:  C[M,N] = scale * (A[M,K] @ Bt[N,K]^T) (+ bias)
// 128 threads = 4 wave32, block tile 128x128, wave tile 64x64 (4x4 WMMA
// tiles -> 16 v_wmma per K-step against 16 fragment ds_loads), K-step 32.
// Async path: 3-stage pipeline via GLOBAL_LOAD_ASYNC_TO_LDS_B128; steady-
// state loop is branch-free (last two iterations peeled, rotating buffer
// indices instead of %3).
// ---------------------------------------------------------------------------
#define GTILE 128
#define GKT    32
#define LSTR   40   // padded LDS row stride (80B) -> conflict-free b128 reads
#if HAVE_ASYNC_LDS
#define NBUF 3
#else
#define NBUF 1
#endif

template <bool HAS_BIAS, bool OUT_BF16>
__global__ __launch_bounds__(128)
void gemm_wmma_bf16(const unsigned short* __restrict__ A, long long strideA,
                    const unsigned short* __restrict__ Bt, long long strideB,
                    const float* __restrict__ bias,
                    void* __restrict__ C, long long strideC,
                    int M, int N, int K, float scale) {
    __shared__ __align__(16) unsigned short lA[NBUF][GTILE][LSTR];
    __shared__ __align__(16) unsigned short lB[NBUF][GTILE][LSTR];

    const int tid  = threadIdx.x;
    const int lane = tid & 31;
    const int wave = tid >> 5;            // 0..3
    const int mblk = blockIdx.y * GTILE;
    const int nblk = blockIdx.x * GTILE;

    A  += (long long)blockIdx.z * strideA;
    Bt += (long long)blockIdx.z * strideB;

    const int waveM = (wave >> 1) * 64;   // 2 waves along M
    const int waveN = (wave & 1) * 64;    // 2 waves along N

    v8f zero = {0.f, 0.f, 0.f, 0.f, 0.f, 0.f, 0.f, 0.f};
    v8f acc[4][4];
#pragma unroll
    for (int i = 0; i < 4; ++i)
#pragma unroll
        for (int j = 0; j < 4; ++j) acc[i][j] = zero;

    // staging map: 128 threads, thread t stages full row t (32 bf16 = 4x b128)
    const int srow = tid;

    // fragment map (16-bit A layout: lanes 0-15 K{0..7,16..23}, 16-31 K{8..15,24..31})
    const int fr = lane & 15;
    const int fk = (lane >> 4) * 8;

    const int ntiles = K / GKT;

#if HAVE_ASYNC_LDS
    auto stage = [&](int buf, int kt) {
        const unsigned short* ga = A + (long long)(mblk + srow) * K + kt;
        const unsigned short* gb = Bt + (long long)(nblk + srow) * K + kt;
#pragma unroll
        for (int c = 0; c < 32; c += 8) {
            async_ld16(ga + c, &lA[buf][srow][c]);
            async_ld16(gb + c, &lB[buf][srow][c]);
        }
    };
#else
    auto stage_sync = [&](int kt) {
        const unsigned short* ga = A + (long long)(mblk + srow) * K + kt;
        const unsigned short* gb = Bt + (long long)(nblk + srow) * K + kt;
        ush8 av[4], bv[4];
#pragma unroll
        for (int c = 0; c < 4; ++c) av[c] = *(const ush8*)(ga + c * 8);
#pragma unroll
        for (int c = 0; c < 4; ++c) bv[c] = *(const ush8*)(gb + c * 8);
        __builtin_prefetch(ga + GKT, 0, 1);
        __builtin_prefetch(gb + GKT, 0, 1);
#pragma unroll
        for (int c = 0; c < 4; ++c) *(ush8*)&lA[0][srow][c * 8] = av[c];
#pragma unroll
        for (int c = 0; c < 4; ++c) *(ush8*)&lB[0][srow][c * 8] = bv[c];
    };
#endif

    auto compute = [&](int buf) {
        const unsigned short (*cA)[LSTR] = lA[buf];
        const unsigned short (*cB)[LSTR] = lB[buf];
        v16bf af[4], bf[4];
#pragma unroll
        for (int mi = 0; mi < 4; ++mi) {
            const int r = waveM + mi * 16 + fr;
            ush8 lo = *(const ush8*)&cA[r][fk];
            ush8 hi = *(const ush8*)&cA[r][fk + 16];
            af[mi] = __builtin_bit_cast(
                v16bf, __builtin_shufflevector(lo, hi, 0, 1, 2, 3, 4, 5, 6, 7, 8,
                                               9, 10, 11, 12, 13, 14, 15));
        }
#pragma unroll
        for (int ni = 0; ni < 4; ++ni) {
            const int r = waveN + ni * 16 + fr;
            ush8 lo = *(const ush8*)&cB[r][fk];
            ush8 hi = *(const ush8*)&cB[r][fk + 16];
            bf[ni] = __builtin_bit_cast(
                v16bf, __builtin_shufflevector(lo, hi, 0, 1, 2, 3, 4, 5, 6, 7, 8,
                                               9, 10, 11, 12, 13, 14, 15));
        }
#pragma unroll
        for (int mi = 0; mi < 4; ++mi)
#pragma unroll
            for (int ni = 0; ni < 4; ++ni)
                acc[mi][ni] = __builtin_amdgcn_wmma_f32_16x16x32_bf16(
                    false, af[mi], false, bf[ni], (short)0, acc[mi][ni],
                    false, false);
    };

#if HAVE_ASYNC_LDS
    stage(0, 0);
    if (ntiles > 1) stage(1, GKT);

    int cur = 0, stg = 2;
    // steady state: branch-free (wait for tile `it`: <= next tile's 8 copies)
    for (int it = 0; it + 2 < ntiles; ++it) {
        WAIT_ASYNC(8);
        __syncthreads();                     // everyone done with buf[stg]
        stage(stg, (it + 2) * GKT);
        compute(cur);
        cur = (cur == 2) ? 0 : cur + 1;
        stg = (stg == 2) ? 0 : stg + 1;
    }
    if (ntiles > 1) {                        // it = ntiles-2: nothing to stage
        WAIT_ASYNC(8);
        __syncthreads();
        compute(cur);
        cur = (cur == 2) ? 0 : cur + 1;
    }
    WAIT_ASYNC(0);                           // drain: last tile fully in LDS
    __syncthreads();
    compute(cur);
#else
    for (int it = 0; it < ntiles; ++it) {
        stage_sync(it * GKT);
        __syncthreads();
        compute(0);
        __syncthreads();
    }
#endif

    // ---- epilogue: C/D layout = VGPR v -> row v (lanes 0-15) / v+8 (16-31) ----
    const int crow0 = mblk + waveM + ((lane >> 4) << 3);
    const int ccol  = nblk + waveN + (lane & 15);
    const long long cbase = (long long)blockIdx.z * strideC;
#pragma unroll
    for (int mi = 0; mi < 4; ++mi) {
#pragma unroll
        for (int ni = 0; ni < 4; ++ni) {
            const int col = ccol + ni * 16;
            float badd = HAS_BIAS ? bias[col] : 0.0f;
#pragma unroll
            for (int v = 0; v < 8; ++v) {
                const int row = crow0 + mi * 16 + v;
                const float val = acc[mi][ni][v] * scale + badd;
                if (OUT_BF16)
                    ((unsigned short*)C)[cbase + (long long)row * N + col] =
                        f32_to_bf16(val);
                else
                    ((float*)C)[cbase + (long long)row * N + col] = val;
            }
        }
    }
}

// ---------------------------------------------------------------------------
// Row softmax (f32 in, bf16 probabilities out). One 256-thread block per row.
// ---------------------------------------------------------------------------
__global__ __launch_bounds__(256)
void softmax_rows_bf16(const float* __restrict__ S, unsigned short* __restrict__ P,
                       int n) {
    const long long row = blockIdx.x;
    const float* src = S + row * n;
    unsigned short* dst = P + row * n;
    __shared__ float red[256];

    float lmax = -3.4e38f;
    for (int i = threadIdx.x; i < n; i += 256) lmax = fmaxf(lmax, src[i]);
    red[threadIdx.x] = lmax;
    __syncthreads();
    for (int s = 128; s > 0; s >>= 1) {
        if (threadIdx.x < s)
            red[threadIdx.x] = fmaxf(red[threadIdx.x], red[threadIdx.x + s]);
        __syncthreads();
    }
    const float rmax = red[0];
    __syncthreads();

    float lsum = 0.f;
    for (int i = threadIdx.x; i < n; i += 256) lsum += __expf(src[i] - rmax);
    red[threadIdx.x] = lsum;
    __syncthreads();
    for (int s = 128; s > 0; s >>= 1) {
        if (threadIdx.x < s) red[threadIdx.x] += red[threadIdx.x + s];
        __syncthreads();
    }
    const float inv = 1.0f / red[0];

    for (int i = threadIdx.x; i < n; i += 256)
        dst[i] = f32_to_bf16(__expf(src[i] - rmax) * inv);
}

// ---------------------------------------------------------------------------
// Host-side orchestration
// ---------------------------------------------------------------------------
extern "C" void kernel_launch(void* const* d_in, const int* in_sizes, int n_in,
                              void* d_out, int out_size, void* d_ws, size_t ws_size,
                              hipStream_t stream) {
    (void)in_sizes; (void)n_in; (void)out_size; (void)ws_size;
    constexpr int B = 4, S = 2048, D = 1024;

    const float* x  = (const float*)d_in[0];
    const float* Wq = (const float*)d_in[1];
    const float* bq = (const float*)d_in[2];
    const float* Wk = (const float*)d_in[3];
    const float* bk = (const float*)d_in[4];
    const float* Wv = (const float*)d_in[5];
    const float* bv = (const float*)d_in[6];
    float* out = (float*)d_out;

    // carve workspace (256B aligned)
    char* p = (char*)d_ws;
    auto take = [&](size_t bytes) -> char* {
        char* r = p;
        p += (bytes + 255) & ~(size_t)255;
        return r;
    };
    unsigned short* xh  = (unsigned short*)take((size_t)B * S * D * 2);
    unsigned short* wqh = (unsigned short*)take((size_t)D * D * 2);
    unsigned short* wkh = (unsigned short*)take((size_t)D * D * 2);
    unsigned short* wvh = (unsigned short*)take((size_t)D * D * 2);
    unsigned short* qh  = (unsigned short*)take((size_t)B * S * D * 2);
    unsigned short* kh  = (unsigned short*)take((size_t)B * S * D * 2);
    unsigned short* vh  = (unsigned short*)take((size_t)B * S * D * 2);
    unsigned short* vt  = (unsigned short*)take((size_t)B * S * D * 2);
    float*          sc  = (float*)take((size_t)B * S * S * 4);
    unsigned short* pbf = (unsigned short*)take((size_t)B * S * S * 2);

    // 1) f32 -> bf16 conversions
    cvt_f32_bf16<<<(B * S * D) / 1024, 256, 0, stream>>>(x, xh, (long long)B * S * D);
    cvt_f32_bf16<<<(D * D) / 1024, 256, 0, stream>>>(Wq, wqh, (long long)D * D);
    cvt_f32_bf16<<<(D * D) / 1024, 256, 0, stream>>>(Wk, wkh, (long long)D * D);
    cvt_f32_bf16<<<(D * D) / 1024, 256, 0, stream>>>(Wv, wvh, (long long)D * D);

    const long long sSD = (long long)S * D;
    const long long sSS = (long long)S * S;

    // 2) QKV projections: y = x @ W^T + b  (Bt = W rows directly)
    dim3 gProj(D / GTILE, S / GTILE, B);
    gemm_wmma_bf16<true, true><<<gProj, 128, 0, stream>>>(
        xh, sSD, wqh, 0, bq, qh, sSD, S, D, D, 1.0f);
    gemm_wmma_bf16<true, true><<<gProj, 128, 0, stream>>>(
        xh, sSD, wkh, 0, bk, kh, sSD, S, D, D, 1.0f);
    gemm_wmma_bf16<true, true><<<gProj, 128, 0, stream>>>(
        xh, sSD, wvh, 0, bv, vh, sSD, S, D, D, 1.0f);

    // 3) transpose V -> Vt[D,S] so P@V uses the fast non-transposed path
    dim3 gT(D / 32, S / 32, B);
    transpose_bf16<<<gT, 256, 0, stream>>>(vh, vt, S, D);

    // 4) scores = (Q @ K^T) / sqrt(D)   (Bt = K rows)
    dim3 gScore(S / GTILE, S / GTILE, B);
    gemm_wmma_bf16<false, false><<<gScore, 128, 0, stream>>>(
        qh, sSD, kh, sSD, nullptr, sc, sSS, S, S, D, 0.03125f);

    // 5) softmax rows -> bf16 probabilities
    softmax_rows_bf16<<<B * S, 256, 0, stream>>>(sc, pbf, S);

    // 6) out = P @ V = Pbf @ Vt^T   (Bt = Vt rows)
    dim3 gAV(D / GTILE, S / GTILE, B);
    gemm_wmma_bf16<false, false><<<gAV, 128, 0, stream>>>(
        pbf, sSS, vt, sSD, nullptr, out, sSD, S, D, S, 1.0f);
}
